// GraphEncoder_GAT_10917806866967
// MI455X (gfx1250) — matrix operand order repaired and checked by reading, not compiled, not run on hardware
//
#include <hip/hip_runtime.h>
#include <hip/hip_bf16.h>
#include <math.h>

#define NN 50000
#define EE 800000
#define EP 850000   /* EE + NN self-loops */
#define HH 4
#define CC 32
#define HC 128
#define NHID 512
#define NOUT 768
#define GG 256
#define NEG_SLOPE 0.2f

typedef __attribute__((ext_vector_type(16))) _Float16 v16h;
typedef __attribute__((ext_vector_type(8)))  float    v8f;

// Monotone order-preserving float<->uint encoding for atomicMax over any-sign floats.
__device__ __forceinline__ unsigned enc_f32(float f) {
  unsigned u = __float_as_uint(f);
  return (u & 0x80000000u) ? ~u : (u | 0x80000000u);
}
__device__ __forceinline__ float dec_f32(unsigned u) {
  return (u & 0x80000000u) ? __uint_as_float(u ^ 0x80000000u) : __uint_as_float(~u);
}

// Async global->LDS copy of one dword per lane (GV mode, ASYNCcnt-tracked).
__device__ __forceinline__ void async_g2lds_b32(unsigned lds_byte_off, const float* gaddr) {
  asm volatile("global_load_async_to_lds_b32 %0, %1, off"
               :: "v"(lds_byte_off), "v"(gaddr) : "memory");
}
__device__ __forceinline__ void wait_asynccnt0() {
  asm volatile("s_wait_asynccnt 0" ::: "memory");
}

// C[M,Nd] = A[M,K] @ B[K,Nd] (+bias, optional relu).
// Block = 256 threads = 8 waves; blockIdx.y picks a 16-wide N panel, each wave
// owns one 16x16 tile (8 consecutive M tiles per block). The B panel (K x 16,
// f32) is staged once per block into LDS, [n][k]-major with +4 dword pad,
// via global_load_async_to_lds_b32 (transposing scatter on the LDS address).
// K is a template parameter so the WMMA chain fully unrolls (K/32 v_wmma ops).
template <int K>
__global__ __launch_bounds__(256) void gemm_f16_wmma(
    const float* __restrict__ A, const float* __restrict__ B,
    const float* __restrict__ bias, float* __restrict__ C,
    int M, int Nd, int relu)
{
  constexpr int KP = K + 4;              // padded K stride (floats) for LDS banking
  __shared__ float bpan[16 * KP];

  const int tn = blockIdx.y;

  // Cooperative async transposing copy: B[k][tn*16+j] -> bpan[j*KP + k].
  for (int idx = threadIdx.x; idx < K * 16; idx += 256) {
    int k = idx >> 4, j = idx & 15;
    unsigned ldsoff = (unsigned)(uintptr_t)&bpan[j * KP + k];
    async_g2lds_b32(ldsoff, B + (size_t)k * Nd + tn * 16 + j);
  }
  wait_asynccnt0();
  __syncthreads();

  const int wave = threadIdx.x >> 5;
  const int tm = blockIdx.x * 8 + wave;
  if (tm * 16 >= M) return;
  const int lane = threadIdx.x & 31;
  const int r  = lane & 15;              // A row-in-tile / B col-in-tile
  const int hi = lane >> 4;              // lane-half selects K sub-group (ISA layout)

  v8f c = {};
#pragma unroll
  for (int k0 = 0; k0 < K; k0 += 32) {
    // A fragment (16x32 f16): lane holds row m = tm*16+r;
    //   VGPR v in 0..3 -> K = k0 + hi*8 + 2v{,+1}; v in 4..7 -> +16.
    const float* arow = A + (size_t)(tm * 16 + r) * K + (k0 + hi * 8);
    // B fragment (32x16 f16): lane holds col n; element i -> K = k0 + hi*8 + i
    // for i < 8, K = k0 + 16 + hi*8 + (i-8) for i >= 8 (same as A).
    const float* bl = &bpan[r * KP + k0 + hi * 8];
    v16h a, b;
#pragma unroll
    for (int v = 0; v < 4; ++v) {
      a[2 * v]         = (_Float16)arow[2 * v];
      a[2 * v + 1]     = (_Float16)arow[2 * v + 1];
      a[8 + 2 * v]     = (_Float16)arow[16 + 2 * v];
      a[8 + 2 * v + 1] = (_Float16)arow[16 + 2 * v + 1];
    }
#pragma unroll
    for (int i = 0; i < 16; ++i)
      b[i] = (_Float16)bl[i < 8 ? i : i + 8];
    c = __builtin_amdgcn_wmma_f32_16x16x32_f16(false, a, false, b, (short)0, c, false, false);
  }

  // C/D layout: VGPR v -> m = v + 8*hi; col n = tn*16 + r.
  const int ncol = tn * 16 + r;
  const float bv = bias ? bias[ncol] : 0.0f;
#pragma unroll
  for (int v = 0; v < 8; ++v) {
    float val = c[v] + bv;
    if (relu) val = fmaxf(val, 0.0f);
    C[(size_t)(tm * 16 + hi * 8 + v) * Nd + ncol] = val;
  }
}

// Per-(node,head) attention coefficients: e_src/e_dst = <h[n,head,:], a_src/a_dst>.
__global__ void attn_coef(const float* __restrict__ h, const float* __restrict__ as,
                          const float* __restrict__ ad, float* __restrict__ esrc,
                          float* __restrict__ edst)
{
  int t = blockIdx.x * blockDim.x + threadIdx.x;
  if (t >= NN * HH) return;
  int n = t >> 2, hd = t & 3;
  const float* row = h + (size_t)n * HC + hd * CC;
  float s0 = 0.f, s1 = 0.f;
#pragma unroll
  for (int c = 0; c < CC; ++c) {
    float v = row[c];
    s0 += v * as[hd * CC + c];
    s1 += v * ad[hd * CC + c];
  }
  esrc[t] = s0;
  edst[t] = s1;
}

__device__ __forceinline__ void edge_endpoints(const int* __restrict__ ei, int e,
                                               int& s, int& d) {
  if (e < EE) { s = ei[e]; d = ei[EE + e]; }
  else        { s = e - EE; d = e - EE; }       // appended self-loops
}

// Pass 1: segment max of leaky_relu(e_src[src]+e_dst[dst]) over dst, per head.
__global__ void edge_max_k(const int* __restrict__ ei, const float* __restrict__ esrc,
                           const float* __restrict__ edst, unsigned* __restrict__ menc)
{
  int t = blockIdx.x * blockDim.x + threadIdx.x;
  if (t >= EP * HH) return;
  int e = t >> 2, hd = t & 3;
  int s, d; edge_endpoints(ei, e, s, d);
  float v = esrc[s * HH + hd] + edst[d * HH + hd];
  v = (v > 0.f) ? v : NEG_SLOPE * v;
  atomicMax(&menc[d * HH + hd], enc_f32(v));
}

// Pass 2: z[dst] = sum exp(e - m[dst]).
__global__ void edge_sum_k(const int* __restrict__ ei, const float* __restrict__ esrc,
                           const float* __restrict__ edst, const unsigned* __restrict__ menc,
                           float* __restrict__ z)
{
  int t = blockIdx.x * blockDim.x + threadIdx.x;
  if (t >= EP * HH) return;
  int e = t >> 2, hd = t & 3;
  int s, d; edge_endpoints(ei, e, s, d);
  float v = esrc[s * HH + hd] + edst[d * HH + hd];
  v = (v > 0.f) ? v : NEG_SLOPE * v;
  atomicAdd(&z[d * HH + hd], expf(v - dec_f32(menc[d * HH + hd])));
}

// Pass 3: acc[dst,:] += alpha * h[src,:]. One wave per edge; lane owns 4 features.
__global__ void edge_aggr_k(const int* __restrict__ ei, const float* __restrict__ esrc,
                            const float* __restrict__ edst, const unsigned* __restrict__ menc,
                            const float* __restrict__ z, const float* __restrict__ h,
                            float* __restrict__ acc)
{
  long long t = (long long)blockIdx.x * blockDim.x + threadIdx.x;
  if (t >= (long long)EP * 32) return;
  int e = (int)(t >> 5);
  int lane = (int)(t & 31);
  int s, d; edge_endpoints(ei, e, s, d);
  int hd = lane >> 3;                          // features 4*lane..4*lane+3 -> head lane/8
  float v = esrc[s * HH + hd] + edst[d * HH + hd];
  v = (v > 0.f) ? v : NEG_SLOPE * v;
  float alpha = expf(v - dec_f32(menc[d * HH + hd])) / z[d * HH + hd];
  float4 hv = ((const float4*)(h + (size_t)s * HC))[lane];   // L2-resident gather
  float* o = acc + (size_t)d * HC + lane * 4;
  atomicAdd(o + 0, alpha * hv.x);
  atomicAdd(o + 1, alpha * hv.y);
  atomicAdd(o + 2, alpha * hv.z);
  atomicAdd(o + 3, alpha * hv.w);
}

// x_next = relu(skip + acc + b)   (skip already has its own bias)
__global__ void combine_relu(const float* __restrict__ skip, const float* __restrict__ acc,
                             const float* __restrict__ b, float* __restrict__ xout)
{
  int t = blockIdx.x * blockDim.x + threadIdx.x;
  if (t >= NN * HC) return;
  int f = t & (HC - 1);
  xout[t] = fmaxf(skip[t] + acc[t] + b[f], 0.0f);
}

// Global max-pool over batch; values are post-ReLU (>=0) so signed-int bit compare works.
__global__ void pool_max(const float* __restrict__ xcur, const int* __restrict__ batch,
                         float* __restrict__ pooled)
{
  int t = blockIdx.x * blockDim.x + threadIdx.x;
  if (t >= NN * HC) return;
  int n = t >> 7, f = t & 127;
  int g = batch[n];
  atomicMax((int*)&pooled[(size_t)g * HC + f], __float_as_int(xcur[t]));
}

template <int K>
static void launch_gemm(const float* A, const float* B, const float* bias, float* C,
                        int M, int Nd, int relu, hipStream_t stream)
{
  int mtiles = M >> 4;
  dim3 grid((mtiles + 7) / 8, Nd >> 4);
  gemm_f16_wmma<K><<<grid, 256, 0, stream>>>(A, B, bias, C, M, Nd, relu);
}

extern "C" void kernel_launch(void* const* d_in, const int* in_sizes, int n_in,
                              void* d_out, int out_size, void* d_ws, size_t ws_size,
                              hipStream_t stream)
{
  const float* x     = (const float*)d_in[0];
  const int*   ei    = (const int*)d_in[1];
  const int*   batch = (const int*)d_in[2];
  const float* W[3]   = {(const float*)d_in[3],  (const float*)d_in[9],  (const float*)d_in[15]};
  const float* As[3]  = {(const float*)d_in[4],  (const float*)d_in[10], (const float*)d_in[16]};
  const float* Ad[3]  = {(const float*)d_in[5],  (const float*)d_in[11], (const float*)d_in[17]};
  const float* Bb[3]  = {(const float*)d_in[6],  (const float*)d_in[12], (const float*)d_in[18]};
  const float* SkW[3] = {(const float*)d_in[7],  (const float*)d_in[13], (const float*)d_in[19]};
  const float* SkB[3] = {(const float*)d_in[8],  (const float*)d_in[14], (const float*)d_in[20]};
  const float* m1w = (const float*)d_in[21];
  const float* m1b = (const float*)d_in[22];
  const float* m2w = (const float*)d_in[23];
  const float* m2b = (const float*)d_in[24];
  float* out = (float*)d_out;

  // Workspace layout (~102 MB; entire working set fits in 192 MB L2).
  float* ws = (float*)d_ws;
  size_t off = 0;
  float*    h      = ws + off; off += (size_t)NN * HC;
  float*    skip   = ws + off; off += (size_t)NN * HC;
  float*    acc    = ws + off; off += (size_t)NN * HC;
  float*    xcur   = ws + off; off += (size_t)NN * HC;
  float*    esrc   = ws + off; off += (size_t)NN * HH;
  float*    edst   = ws + off; off += (size_t)NN * HH;
  unsigned* menc   = (unsigned*)(ws + off); off += (size_t)NN * HH;
  float*    z      = ws + off; off += (size_t)NN * HH;
  float*    pooled = ws + off; off += (size_t)GG * HC;
  float*    hid    = ws + off; off += (size_t)GG * NHID;

  const int TB = 256;

  for (int l = 0; l < 3; ++l) {
    const float* xin = (l == 0) ? x : xcur;
    launch_gemm<HC>(xin, W[l],   nullptr, h,    NN, HC, 0, stream);
    launch_gemm<HC>(xin, SkW[l], SkB[l],  skip, NN, HC, 0, stream);
    attn_coef<<<(NN * HH + TB - 1) / TB, TB, 0, stream>>>(h, As[l], Ad[l], esrc, edst);
    hipMemsetAsync(menc, 0, (size_t)NN * HH * sizeof(unsigned), stream);
    hipMemsetAsync(z,    0, (size_t)NN * HH * sizeof(float), stream);
    hipMemsetAsync(acc,  0, (size_t)NN * HC * sizeof(float), stream);
    edge_max_k<<<(EP * HH + TB - 1) / TB, TB, 0, stream>>>(ei, esrc, edst, menc);
    edge_sum_k<<<(EP * HH + TB - 1) / TB, TB, 0, stream>>>(ei, esrc, edst, menc, z);
    long long tagg = (long long)EP * 32;
    edge_aggr_k<<<(int)((tagg + TB - 1) / TB), TB, 0, stream>>>(ei, esrc, edst, menc, z, h, acc);
    combine_relu<<<(NN * HC + TB - 1) / TB, TB, 0, stream>>>(skip, acc, Bb[l], xcur);
  }

  hipMemsetAsync(pooled, 0, (size_t)GG * HC * sizeof(float), stream);
  pool_max<<<(NN * HC + TB - 1) / TB, TB, 0, stream>>>(xcur, batch, pooled);

  launch_gemm<HC>  (pooled, m1w, m1b, hid, GG, NHID, 1, stream);
  launch_gemm<NHID>(hid,    m2w, m2b, out, GG, NOUT, 0, stream);
}
